// DevignBaseline_2602750181530
// MI455X (gfx1250) — matrix operand order repaired and checked by reading, not compiled, not run on hardware
//
#include <hip/hip_runtime.h>
#include <hip/hip_bf16.h>

// ---------------------------------------------------------------------------
// Devign GCN on gfx1250: bf16 WMMA GEMMs (fp32 accum), packed-B in LDS,
// fp32 edge scatter with float4 gathers.
// ---------------------------------------------------------------------------

typedef __attribute__((ext_vector_type(16))) __bf16 v16bf;
typedef __attribute__((ext_vector_type(8)))  float  v8f;

#define N_NODES 40000
#define N_EDGES 640000
#define N_GRAPHS 64
#define D_IN 128
#define HID 200
#define HID4 50      // HID / 4 (float4 chunks)
#define HIDP_N 208   // HID padded to 16 (WMMA N tiles)
#define HIDP_K 224   // HID padded to 32 (WMMA K steps)
#define C1 100       // HID/2
#define C1P_N 112    // padded to 16
#define C1P_K 128    // padded to 32
#define OUTC 2
#define OUTP_N 16

__device__ __forceinline__ unsigned short f2bf(float f) {
  unsigned int u = __float_as_uint(f);
  u += 0x7FFFu + ((u >> 16) & 1u);   // round-to-nearest-even
  return (unsigned short)(u >> 16);
}

// -------- generic f32 -> zero-padded bf16 (A-side: x, hidden) --------------
__global__ void k_pad_cvt_bf16(const float* __restrict__ src, int M, int N,
                               unsigned short* __restrict__ dst, int Mp, int Np) {
  int i = blockIdx.x * blockDim.x + threadIdx.x;
  int total = Mp * Np;
  if (i >= total) return;
  int r = i / Np, c = i % Np;
  float v = (r < M && c < N) ? src[r * N + c] : 0.0f;
  dst[i] = f2bf(v);
}

// -------- pack f32 weight [K x N] into WMMA B tiles (contiguous k order) -----------------
// Bp flat uint index i: v = i&7, lane = (i>>3)&31, tile = i>>8 = ks*ntiles+nt
// lane half (lane>=16) selects K+16; dword v holds K rows (kb+2v, kb+2v+1).
__global__ void k_pack_w(const float* __restrict__ W, int K, int N,
                         unsigned int* __restrict__ Bp, int ksteps, int ntiles) {
  int i = blockIdx.x * blockDim.x + threadIdx.x;
  int total = ksteps * ntiles * 256;
  if (i >= total) return;
  int v = i & 7;
  int lane = (i >> 3) & 31;
  int tile = i >> 8;
  int nt = tile % ntiles, ks = tile / ntiles;
  int col = nt * 16 + (lane & 15);
  int kb = ks * 32 + (lane >> 4) * 16 + 2 * v;
  float lo = (kb     < K && col < N) ? W[kb * N + col]       : 0.0f;
  float hi = (kb + 1 < K && col < N) ? W[(kb + 1) * N + col] : 0.0f;
  Bp[i] = (unsigned int)f2bf(lo) | ((unsigned int)f2bf(hi) << 16);
}

__global__ void k_fill_f32(float* __restrict__ p, int n, float v) {
  int i = blockIdx.x * blockDim.x + threadIdx.x;
  if (i < n) p[i] = v;
}

__global__ void k_edge_deg(const int* __restrict__ dst, int E,
                           float* __restrict__ deg) {
  int e = blockIdx.x * blockDim.x + threadIdx.x;
  if (e < E) atomicAdd(&deg[dst[e]], 1.0f);
}

__global__ void k_rsqrt_inplace(float* __restrict__ p, int n) {
  int i = blockIdx.x * blockDim.x + threadIdx.x;
  if (i < n) p[i] = rsqrtf(fmaxf(p[i], 1.0f));
}

// -------- WMMA bf16 GEMM: C[M x N] = A[M x K] * Bp (+bias, relu) -----------
// A: bf16 row-major, lda (K zero-padded, K%32==0). Bp: packed tiles (above),
// staged through LDS. One wave computes FOUR 16x16 m-tiles at one n-tile.
// Requires M%64==0, Npad%16==0 -> all loads unmasked, EXEC all ones.
__global__ void k_gemm_bf16_wmma(const unsigned short* __restrict__ A, int lda,
                                 const unsigned int* __restrict__ Bp,
                                 float* __restrict__ C, int ldc,
                                 int M, int Npad, int N, int K,
                                 const float* __restrict__ bias, int act) {
  extern __shared__ unsigned int lds_b[];
  const int ntiles = Npad >> 4;
  const int ksteps = K >> 5;
  const int tid = threadIdx.x;

  // cooperative: stage entire packed weight matrix into LDS
  {
    const int nQ = ksteps * ntiles * 64;           // uint4 count
    const uint4* gq = (const uint4*)Bp;
    uint4* lq = (uint4*)lds_b;
    for (int i = tid; i < nQ; i += blockDim.x) lq[i] = gq[i];
  }
  __syncthreads();

  const int wave = blockIdx.x * (blockDim.x >> 5) + (tid >> 5);
  const int lane = tid & 31;
  const int mgroups = M >> 6;                      // groups of 4 m-tiles
  if (wave >= mgroups * ntiles) return;
  const int mg = wave / ntiles;
  const int nt = wave % ntiles;
  const int half = lane >> 4;
  const int l16  = lane & 15;
  const int col  = nt * 16 + l16;

  union AF { v16bf v; uint4 q[2]; };
  union { v16bf v; uint4 q[2]; } b;
  union CF { v8f v; float f[8]; } acc[4];
#pragma unroll
  for (int g = 0; g < 4; ++g)
#pragma unroll
    for (int i = 0; i < 8; ++i) acc[g].f[i] = 0.0f;

  const unsigned short* Abase = A + (long)(mg * 64 + l16) * lda + half * 16;
  for (int ks = 0; ks < ksteps; ++ks) {
    const int k0 = ks * 32;
    const unsigned int* bl = lds_b + (((ks * ntiles + nt) * 32 + lane) << 3);
    b.q[0] = *(const uint4*)(bl);
    b.q[1] = *(const uint4*)(bl + 4);
    if (ks + 1 < ksteps) __builtin_prefetch(Abase + k0 + 32, 0, 3);
#pragma unroll
    for (int g = 0; g < 4; ++g) {
      AF a;
      const unsigned short* ap = Abase + (long)g * 16 * lda + k0;
      a.q[0] = *(const uint4*)(ap);
      a.q[1] = *(const uint4*)(ap + 8);
      acc[g].v = __builtin_amdgcn_wmma_f32_16x16x32_bf16(
          /*neg_a=*/false, a.v, /*neg_b=*/false, b.v,
          /*c_mod=*/(short)0, acc[g].v, /*reuse_a=*/false, /*reuse_b=*/false);
    }
  }

  if (col < N) {
    const float bv = bias ? bias[col] : 0.0f;
#pragma unroll
    for (int g = 0; g < 4; ++g) {
#pragma unroll
      for (int r = 0; r < 8; ++r) {
        int row = mg * 64 + g * 16 + r + half * 8;
        float v = acc[g].f[r] + bv;
        if (act) v = fmaxf(v, 0.0f);
        C[(long)row * ldc + col] = v;
      }
    }
  }
}

// -------- self-loop term: out = dinv^2 * t (float4) ------------------------
__global__ void k_selfloop(const float* __restrict__ t,
                           const float* __restrict__ dinv,
                           float* __restrict__ out, int n4 /* N_NODES*HID4 */) {
  int i = blockIdx.x * blockDim.x + threadIdx.x;
  if (i >= n4) return;
  int node = i / HID4;
  float d = dinv[node];
  float w = d * d;
  float4 v = ((const float4*)t)[i];
  v.x *= w; v.y *= w; v.z *= w; v.w *= w;
  ((float4*)out)[i] = v;
}

// -------- edge scatter: one wave per edge, float4 gather + f32 atomics -----
__global__ void k_edge_agg(const int* __restrict__ src, const int* __restrict__ dst,
                           const float* __restrict__ dinv,
                           const float* __restrict__ t, float* __restrict__ out,
                           int E) {
  int wave = (blockIdx.x * blockDim.x + threadIdx.x) >> 5;
  int lane = threadIdx.x & 31;
  if (wave >= E) return;
  int s = src[wave], d = dst[wave];
  float w = dinv[s] * dinv[d];
  const float4* ts = (const float4*)(t + (long)s * HID);
  float* od = out + (long)d * HID;
#pragma unroll
  for (int c = 0; c < 2; ++c) {
    int f4 = lane + c * 32;
    if (f4 < HID4) {
      float4 v = ts[f4];
      atomicAdd(&od[f4 * 4 + 0], w * v.x);
      atomicAdd(&od[f4 * 4 + 1], w * v.y);
      atomicAdd(&od[f4 * 4 + 2], w * v.z);
      atomicAdd(&od[f4 * 4 + 3], w * v.w);
    }
  }
}

// -------- bias + relu (fp32, in place) + repack to zero-padded bf16 --------
__global__ void k_bias_relu_cvt(float* __restrict__ h /* M x HID, in place */,
                                const float* __restrict__ bias,
                                unsigned short* __restrict__ hbf /* M x HIDP_K */,
                                int M) {
  int i = blockIdx.x * blockDim.x + threadIdx.x;
  int total = M * HIDP_K;
  if (i >= total) return;
  int r = i / HIDP_K, c = i % HIDP_K;
  float v = 0.0f;
  if (c < HID) {
    v = fmaxf(h[r * HID + c] + bias[c], 0.0f);
    h[r * HID + c] = v;
  }
  hbf[i] = f2bf(v);
}

// -------- gate: g_inout = h * sigmoid(g_inout) -----------------------------
__global__ void k_gate(const float* __restrict__ h, float* __restrict__ g,
                       int n) {
  int i = blockIdx.x * blockDim.x + threadIdx.x;
  if (i >= n) return;
  float x = g[i];
  g[i] = h[i] * (1.0f / (1.0f + __expf(-x)));
}

// -------- mean pool (atomic sum + count), one wave per node ----------------
__global__ void k_pool(const float* __restrict__ hg, const int* __restrict__ batch,
                       float* __restrict__ pooled, float* __restrict__ cnt,
                       int Nn) {
  int wave = (blockIdx.x * blockDim.x + threadIdx.x) >> 5;
  int lane = threadIdx.x & 31;
  if (wave >= Nn) return;
  int b = batch[wave];
  const float4* row = (const float4*)(hg + (long)wave * HID);
  float* pb = pooled + b * HID;
#pragma unroll
  for (int c = 0; c < 2; ++c) {
    int f4 = lane + c * 32;
    if (f4 < HID4) {
      float4 v = row[f4];
      atomicAdd(&pb[f4 * 4 + 0], v.x);
      atomicAdd(&pb[f4 * 4 + 1], v.y);
      atomicAdd(&pb[f4 * 4 + 2], v.z);
      atomicAdd(&pb[f4 * 4 + 3], v.w);
    }
  }
  if (lane == 0) atomicAdd(&cnt[b], 1.0f);
}

__global__ void k_pooldiv_cvt(float* __restrict__ pooled,
                              const float* __restrict__ cnt,
                              unsigned short* __restrict__ p16) {
  int i = blockIdx.x * blockDim.x + threadIdx.x;
  if (i >= N_GRAPHS * HIDP_K) return;
  int r = i / HIDP_K, c = i % HIDP_K;
  float v = 0.0f;
  if (c < HID) {
    v = pooled[r * HID + c] / fmaxf(cnt[r], 1.0f);
    pooled[r * HID + c] = v;
  }
  p16[i] = f2bf(v);
}

// ---------------------------------------------------------------------------
extern "C" void kernel_launch(void* const* d_in, const int* in_sizes, int n_in,
                              void* d_out, int out_size, void* d_ws, size_t ws_size,
                              hipStream_t stream) {
  (void)in_sizes; (void)n_in; (void)out_size; (void)ws_size;
  const float* x     = (const float*)d_in[0];
  const int*   ei    = (const int*)d_in[1];          // [2, E] flat
  const int*   batch = (const int*)d_in[2];
  const float* W1  = (const float*)d_in[3];  const float* b1  = (const float*)d_in[4];
  const float* W2  = (const float*)d_in[5];  const float* b2  = (const float*)d_in[6];
  const float* W3  = (const float*)d_in[7];  const float* b3  = (const float*)d_in[8];
  const float* Wg  = (const float*)d_in[9];  const float* bg  = (const float*)d_in[10];
  const float* Wc1 = (const float*)d_in[11]; const float* bc1 = (const float*)d_in[12];
  const float* Wc2 = (const float*)d_in[13]; const float* bc2 = (const float*)d_in[14];
  const int* esrc = ei;
  const int* edst = ei + N_EDGES;

  // packed-weight tile counts
  const int KS_W1 = D_IN / 32,   NT_HID = HIDP_N / 16;   // 4, 13
  const int KS_HID = HIDP_K / 32;                        // 7
  const int NT_C1 = C1P_N / 16;                          // 7
  const int KS_C2 = C1P_K / 32,  NT_OUT = OUTP_N / 16;   // 4, 1

  // ---- carve workspace (256B aligned) ----
  size_t off = 0;
  char* base = (char*)d_ws;
  auto carve = [&](size_t bytes) -> char* {
    char* p = base + off;
    off += (bytes + 255) & ~(size_t)255;
    return p;
  };
  float* deg   = (float*)carve((size_t)N_NODES * 4);
  float* bufA  = (float*)carve((size_t)N_NODES * HID * 4);
  float* bufB  = (float*)carve((size_t)N_NODES * HID * 4);
  unsigned short* act16 = (unsigned short*)carve((size_t)N_NODES * HIDP_K * 2);
  unsigned short* x16   = act16; // alias: x16 dead before act16 first written
  unsigned int* W1p  = (unsigned int*)carve((size_t)KS_W1  * NT_HID * 256 * 4);
  unsigned int* W2p  = (unsigned int*)carve((size_t)KS_HID * NT_HID * 256 * 4);
  unsigned int* W3p  = (unsigned int*)carve((size_t)KS_HID * NT_HID * 256 * 4);
  unsigned int* Wgp  = (unsigned int*)carve((size_t)KS_HID * NT_HID * 256 * 4);
  unsigned int* Wc1p = (unsigned int*)carve((size_t)KS_HID * NT_C1  * 256 * 4);
  unsigned int* Wc2p = (unsigned int*)carve((size_t)KS_C2  * NT_OUT * 256 * 4);
  float* pooled = (float*)carve((size_t)N_GRAPHS * HID * 4);
  float* cnt    = (float*)carve((size_t)N_GRAPHS * 4);
  unsigned short* pooled16 = (unsigned short*)carve((size_t)N_GRAPHS * HIDP_K * 2);
  float* hidden = (float*)carve((size_t)N_GRAPHS * C1 * 4);
  unsigned short* hidden16 = (unsigned short*)carve((size_t)N_GRAPHS * C1P_K * 2);

  const int TB = 256;
  auto blocks = [](long n, int tb) { return (int)((n + tb - 1) / tb); };

  // ---- weight packing + input conversion ----
  hipLaunchKernelGGL(k_pack_w, dim3(blocks((long)KS_W1 * NT_HID * 256, TB)), dim3(TB), 0, stream,
                     W1, D_IN, HID, W1p, KS_W1, NT_HID);
  hipLaunchKernelGGL(k_pack_w, dim3(blocks((long)KS_HID * NT_HID * 256, TB)), dim3(TB), 0, stream,
                     W2, HID, HID, W2p, KS_HID, NT_HID);
  hipLaunchKernelGGL(k_pack_w, dim3(blocks((long)KS_HID * NT_HID * 256, TB)), dim3(TB), 0, stream,
                     W3, HID, HID, W3p, KS_HID, NT_HID);
  hipLaunchKernelGGL(k_pack_w, dim3(blocks((long)KS_HID * NT_HID * 256, TB)), dim3(TB), 0, stream,
                     Wg, HID, HID, Wgp, KS_HID, NT_HID);
  hipLaunchKernelGGL(k_pack_w, dim3(blocks((long)KS_HID * NT_C1 * 256, TB)), dim3(TB), 0, stream,
                     Wc1, HID, C1, Wc1p, KS_HID, NT_C1);
  hipLaunchKernelGGL(k_pack_w, dim3(blocks((long)KS_C2 * NT_OUT * 256, TB)), dim3(TB), 0, stream,
                     Wc2, C1, OUTC, Wc2p, KS_C2, NT_OUT);
  hipLaunchKernelGGL(k_pad_cvt_bf16, dim3(blocks((long)N_NODES * D_IN, TB)), dim3(TB), 0, stream,
                     x, N_NODES, D_IN, x16, N_NODES, D_IN);

  // ---- degree / dinv ----
  hipLaunchKernelGGL(k_fill_f32, dim3(blocks(N_NODES, TB)), dim3(TB), 0, stream, deg, N_NODES, 1.0f);
  hipLaunchKernelGGL(k_edge_deg, dim3(blocks(N_EDGES, TB)), dim3(TB), 0, stream, edst, N_EDGES, deg);
  hipLaunchKernelGGL(k_rsqrt_inplace, dim3(blocks(N_NODES, TB)), dim3(TB), 0, stream, deg, N_NODES);

  const int wavesMain = (N_NODES / 64) * NT_HID;
  const int gemmBlocksMain = blocks((long)wavesMain, TB / 32);
  const size_t ldsW1  = (size_t)KS_W1  * NT_HID * 1024;
  const size_t ldsHID = (size_t)KS_HID * NT_HID * 1024;
  const size_t ldsC1  = (size_t)KS_HID * NT_C1  * 1024;
  const size_t ldsC2  = (size_t)KS_C2  * NT_OUT * 1024;
  const long nh  = (long)N_NODES * HID;
  const long nh4 = (long)N_NODES * HID4;
  const int edgeBlocks = blocks((long)N_EDGES * 32, TB);

  // ---- layer 1 ----
  hipLaunchKernelGGL(k_gemm_bf16_wmma, dim3(gemmBlocksMain), dim3(TB), ldsW1, stream,
                     x16, D_IN, W1p, bufA, HID,
                     N_NODES, HIDP_N, HID, D_IN, (const float*)nullptr, 0);
  hipLaunchKernelGGL(k_selfloop, dim3(blocks(nh4, TB)), dim3(TB), 0, stream, bufA, deg, bufB, (int)nh4);
  hipLaunchKernelGGL(k_edge_agg, dim3(edgeBlocks), dim3(TB), 0, stream,
                     esrc, edst, deg, bufA, bufB, N_EDGES);
  hipLaunchKernelGGL(k_bias_relu_cvt, dim3(blocks((long)N_NODES * HIDP_K, TB)), dim3(TB), 0, stream,
                     bufB, b1, act16, N_NODES);

  // ---- layers 2 and 3 ----
  const unsigned int* Wp[2] = {W2p, W3p};
  const float* bs[2] = {b2, b3};
  for (int l = 0; l < 2; ++l) {
    hipLaunchKernelGGL(k_gemm_bf16_wmma, dim3(gemmBlocksMain), dim3(TB), ldsHID, stream,
                       act16, HIDP_K, Wp[l], bufA, HID,
                       N_NODES, HIDP_N, HID, HIDP_K, (const float*)nullptr, 0);
    hipLaunchKernelGGL(k_selfloop, dim3(blocks(nh4, TB)), dim3(TB), 0, stream, bufA, deg, bufB, (int)nh4);
    hipLaunchKernelGGL(k_edge_agg, dim3(edgeBlocks), dim3(TB), 0, stream,
                       esrc, edst, deg, bufA, bufB, N_EDGES);
    hipLaunchKernelGGL(k_bias_relu_cvt, dim3(blocks((long)N_NODES * HIDP_K, TB)), dim3(TB), 0, stream,
                       bufB, bs[l], act16, N_NODES);
  }

  // ---- gate: g = h @ Wg + bg ; hg = h * sigmoid(g) (hg -> bufA) ----
  hipLaunchKernelGGL(k_gemm_bf16_wmma, dim3(gemmBlocksMain), dim3(TB), ldsHID, stream,
                     act16, HIDP_K, Wgp, bufA, HID,
                     N_NODES, HIDP_N, HID, HIDP_K, bg, 0);
  hipLaunchKernelGGL(k_gate, dim3(blocks(nh, TB)), dim3(TB), 0, stream, bufB, bufA, (int)nh);

  // ---- mean pool ----
  hipLaunchKernelGGL(k_fill_f32, dim3(blocks(N_GRAPHS * HID, TB)), dim3(TB), 0, stream,
                     pooled, N_GRAPHS * HID, 0.0f);
  hipLaunchKernelGGL(k_fill_f32, dim3(1), dim3(TB), 0, stream, cnt, N_GRAPHS, 0.0f);
  hipLaunchKernelGGL(k_pool, dim3(blocks((long)N_NODES * 32, TB)), dim3(TB), 0, stream,
                     bufA, batch, pooled, cnt, N_NODES);
  hipLaunchKernelGGL(k_pooldiv_cvt, dim3(blocks((long)N_GRAPHS * HIDP_K, TB)), dim3(TB), 0, stream,
                     pooled, cnt, pooled16);

  // ---- classifier ----
  const int wavesC1 = (N_GRAPHS / 64) * NT_C1;
  hipLaunchKernelGGL(k_gemm_bf16_wmma, dim3(blocks((long)wavesC1, TB / 32)), dim3(TB), ldsC1, stream,
                     pooled16, HIDP_K, Wc1p, hidden, C1,
                     N_GRAPHS, C1P_N, C1, HIDP_K, bc1, 1);
  hipLaunchKernelGGL(k_pad_cvt_bf16, dim3(blocks((long)N_GRAPHS * C1P_K, TB)), dim3(TB), 0, stream,
                     hidden, N_GRAPHS, C1, hidden16, N_GRAPHS, C1P_K);
  const int wavesC2 = (N_GRAPHS / 64) * NT_OUT;
  hipLaunchKernelGGL(k_gemm_bf16_wmma, dim3(blocks((long)wavesC2, TB / 32)), dim3(TB), ldsC2, stream,
                     hidden16, C1P_K, Wc2p, (float*)d_out, OUTC,
                     N_GRAPHS, OUTP_N, OUTC, C1P_K, bc2, 0);
}
